// TorchReference_63685775065700
// MI455X (gfx1250) — matrix operand order repaired and checked by reading, not compiled, not run on hardware
//
#include <hip/hip_runtime.h>

// ---------------------------------------------------------------------------
// bf16x3-emulated fp32 GEMM for gfx1250 (MI455X), D = bf16(sa*sb*(A@B))
//   A: 4096x4096 f32 (row major)   B: 4096x6144 f32 (row major)
//   D: 4096x6144 bf16
// Block tile 128x256, K-step 32, 8 waves (wave32) in 2x4; each wave computes
// 64x64 via V_WMMA_F32_16X16X32_BF16 with hi/lo bf16 split:
//   acc += Ahi*Bhi + Ahi*Blo + Alo*Bhi   (~fp32 accuracy at 3x bf16 cost)
// Register double-buffering: tile k+1 global loads issue before the WMMA
// block of tile k, so HBM latency hides behind 48 WMMAs.
// ---------------------------------------------------------------------------

typedef __attribute__((ext_vector_type(4)))  float   f32x4;
typedef __attribute__((ext_vector_type(8)))  float   v8f;
typedef __attribute__((ext_vector_type(4)))  __bf16  bf16x4;
typedef __attribute__((ext_vector_type(8)))  __bf16  v8bf;
typedef __attribute__((ext_vector_type(16))) __bf16  v16bf;

#define GEMM_M 4096
#define GEMM_K 4096
#define GEMM_N 6144

#define TILE_M 128
#define TILE_N 256
#define TILE_K 32
#define LDS_STRIDE 40   // 32 halfwords + 8 pad -> 80B row stride, spreads banks

// Load a 16-element bf16 fragment from LDS as two 16B ds_load_b128 runs.
__device__ __forceinline__ v16bf ld_frag(const __bf16* p, int off0, int off1) {
  v8bf a = *(const v8bf*)(p + off0);
  v8bf b = *(const v8bf*)(p + off1);
  return __builtin_shufflevector(a, b, 0,1,2,3,4,5,6,7,8,9,10,11,12,13,14,15);
}

__global__ __launch_bounds__(256)
void gemm_bf16x3_scaled(const float* __restrict__ A,
                        const float* __restrict__ B,
                        const float* __restrict__ scale_a,
                        const float* __restrict__ scale_b,
                        __bf16* __restrict__ D) {
  // LDS: hi/lo bf16 planes; Bt is the transposed B tile (N-major, K contiguous).
  __shared__ __align__(16) __bf16 Ah [TILE_M * LDS_STRIDE];
  __shared__ __align__(16) __bf16 Al [TILE_M * LDS_STRIDE];
  __shared__ __align__(16) __bf16 Bth[TILE_N * LDS_STRIDE];
  __shared__ __align__(16) __bf16 Btl[TILE_N * LDS_STRIDE];

  const int tid    = threadIdx.x;
  const int lane   = tid & 31;
  const int waveId = tid >> 5;         // 0..7
  const int waveM  = waveId >> 2;      // 0..1 -> 64 rows each
  const int waveN  = waveId & 3;       // 0..3 -> 64 cols each
  const int half   = lane >> 4;        // lane half for WMMA K layout
  const int lrow   = lane & 15;

  const int m0 = blockIdx.y * TILE_M;
  const int n0 = blockIdx.x * TILE_N;

  const float s = scale_a[0] * scale_b[0];   // uniform -> scalar loads

  v8f acc[4][4];
  const v8f vzero = {0.f,0.f,0.f,0.f,0.f,0.f,0.f,0.f};
#pragma unroll
  for (int mi = 0; mi < 4; ++mi)
#pragma unroll
    for (int ni = 0; ni < 4; ++ni)
      acc[mi][ni] = vzero;

  // staging coordinates (constant across K loop)
  const int arow = tid >> 3;            // A: 32 rows per 256-thread pass, 4 passes
  const int akq  = (tid & 7) * 4;       // A: float4 along K
  const int bcol = tid;                 // B: one column of the 32x256 tile/thread

  // ---- register double buffer: preload tile k0 = 0 ----
  f32x4 aReg[4];
  f32x4 bReg[8];
#pragma unroll
  for (int i = 0; i < 4; ++i)
    aReg[i] = *(const f32x4*)(A + (size_t)(m0 + arow + i * 32) * GEMM_K + akq);
#pragma unroll
  for (int q = 0; q < 8; ++q) {
#pragma unroll
    for (int j = 0; j < 4; ++j)
      bReg[q][j] = B[(size_t)(q * 4 + j) * GEMM_N + n0 + bcol];
  }

#pragma unroll 1
  for (int k0 = 0; k0 < GEMM_K; k0 += TILE_K) {
    // ---- convert + store the prefetched tile to LDS ----
#pragma unroll
    for (int i = 0; i < 4; ++i) {
      const int r = arow + i * 32;
      bf16x4 hi, lo;
#pragma unroll
      for (int j = 0; j < 4; ++j) {
        __bf16 h = (__bf16)aReg[i][j];            // v_cvt f32->bf16 (RNE)
        hi[j] = h;
        lo[j] = (__bf16)(aReg[i][j] - (float)h);  // residual
      }
      *(bf16x4*)(Ah + r * LDS_STRIDE + akq) = hi;
      *(bf16x4*)(Al + r * LDS_STRIDE + akq) = lo;
    }
#pragma unroll
    for (int q = 0; q < 8; ++q) {
      bf16x4 hi, lo;
#pragma unroll
      for (int j = 0; j < 4; ++j) {
        __bf16 h = (__bf16)bReg[q][j];
        hi[j] = h;
        lo[j] = (__bf16)(bReg[q][j] - (float)h);
      }
      *(bf16x4*)(Bth + bcol * LDS_STRIDE + q * 4) = hi;   // Bt[n][k..k+3]
      *(bf16x4*)(Btl + bcol * LDS_STRIDE + q * 4) = lo;
    }

    __syncthreads();

    // ---- fragments ----
    // A k order used here (contiguous, not ISA table order): lanes 0-15 hold K 0..15 of row m, lanes 16-31 K 16..31
    const int akr = half * 16;
    v16bf ahi[4], alo[4];
#pragma unroll
    for (int mi = 0; mi < 4; ++mi) {
      const int base = (waveM * 64 + mi * 16 + lrow) * LDS_STRIDE;
      ahi[mi] = ld_frag(Ah, base + akr, base + akr + 8);
      alo[mi] = ld_frag(Al, base + akr, base + akr + 8);
    }
    // B k order used here (same as A, not ISA table order): lanes 0-15 hold K 0..15 of col n, lanes 16-31 K 16..31
    const int bkb = half * 16;
    v16bf bhi[4], blo[4];
#pragma unroll
    for (int ni = 0; ni < 4; ++ni) {
      const int base = (waveN * 64 + ni * 16 + lrow) * LDS_STRIDE + bkb;
      bhi[ni] = ld_frag(Bth, base, base + 8);
      blo[ni] = ld_frag(Btl, base, base + 8);
    }

    // ---- issue next tile's global loads; latency hides behind the WMMAs ----
    const int k1 = k0 + TILE_K;
    if (k1 < GEMM_K) {
#pragma unroll
      for (int i = 0; i < 4; ++i)
        aReg[i] = *(const f32x4*)(A + (size_t)(m0 + arow + i * 32) * GEMM_K + k1 + akq);
#pragma unroll
      for (int q = 0; q < 8; ++q) {
#pragma unroll
        for (int j = 0; j < 4; ++j)
          bReg[q][j] = B[(size_t)(k1 + q * 4 + j) * GEMM_N + n0 + bcol];
      }
      // warm L2 one tile further ahead
      __builtin_prefetch(A + (size_t)(m0 + arow) * GEMM_K + k1 + TILE_K + akq, 0, 0);
      __builtin_prefetch(B + (size_t)(k1 + TILE_K) * GEMM_N + n0 + bcol, 0, 0);
    }

    // ---- bf16x3 WMMA: acc += Ahi*Bhi + Ahi*Blo + Alo*Bhi ----
#pragma unroll
    for (int mi = 0; mi < 4; ++mi) {
#pragma unroll
      for (int ni = 0; ni < 4; ++ni) {
        acc[mi][ni] = __builtin_amdgcn_wmma_f32_16x16x32_bf16(
            false, ahi[mi], false, bhi[ni], (short)0, acc[mi][ni], false, false);
        acc[mi][ni] = __builtin_amdgcn_wmma_f32_16x16x32_bf16(
            false, ahi[mi], false, blo[ni], (short)0, acc[mi][ni], false, false);
        acc[mi][ni] = __builtin_amdgcn_wmma_f32_16x16x32_bf16(
            false, alo[mi], false, bhi[ni], (short)0, acc[mi][ni], false, false);
      }
    }

    __syncthreads();   // LDS safe to overwrite next iteration
  }

  // ---- epilogue: D = bf16(s * acc); C/D layout: VGPR v -> row v (+8 hi half)
  const int rsel = half * 8;
  const int rowBase = m0 + waveM * 64;
  const int colBase = n0 + waveN * 64;
#pragma unroll
  for (int mi = 0; mi < 4; ++mi) {
#pragma unroll
    for (int ni = 0; ni < 4; ++ni) {
      const int c  = colBase + ni * 16 + lrow;
      const int r0 = rowBase + mi * 16 + rsel;
      v8f a = acc[mi][ni];
#pragma unroll
      for (int v = 0; v < 8; ++v) {
        D[(size_t)(r0 + v) * GEMM_N + c] = (__bf16)(s * a[v]);
      }
    }
  }
}

extern "C" void kernel_launch(void* const* d_in, const int* in_sizes, int n_in,
                              void* d_out, int out_size, void* d_ws, size_t ws_size,
                              hipStream_t stream) {
  const float* A  = (const float*)d_in[0];
  const float* B  = (const float*)d_in[1];
  // d_in[2] (C) is unused by the reference
  const float* sa = (const float*)d_in[3];
  const float* sb = (const float*)d_in[4];
  __bf16* D = (__bf16*)d_out;

  dim3 grid(GEMM_N / TILE_N, GEMM_M / TILE_M);  // 24 x 32
  dim3 block(256);
  gemm_bf16x3_scaled<<<grid, block, 0, stream>>>(A, B, sa, sb, D);
}
